// MemoryCell_45878840656430
// MI455X (gfx1250) — matrix operand order repaired and checked by reading, not compiled, not run on hardware
//
#include <hip/hip_runtime.h>
#include <math.h>

// Problem dims (fixed by the reference)
#define BATCH   64
#define SEQLEN  1024
#define DIN     512
#define HGATES  2048
#define H4      512
#define NROWS   (BATCH * SEQLEN)   // 65536 rows of the input-projection GEMM

#define LDSROW  520                // padded bf16 row stride (1040 B, bank-skewed)

typedef __attribute__((ext_vector_type(16))) __bf16        v16bf;
typedef __attribute__((ext_vector_type(8)))  float         v8f;
typedef __attribute__((ext_vector_type(4)))  unsigned int  v4u;

union Frag16 { v16bf v; v4u u[2]; };

__device__ __forceinline__ unsigned short f32_to_bf16_rne(float f) {
  unsigned u = __float_as_uint(f);
  unsigned r = u + 0x7FFFu + ((u >> 16) & 1u);
  return (unsigned short)(r >> 16);
}

// 16-bit A matrix 16x32 (MxK): lane L -> row M=L&15; K chunks at
// k+(L>>4)*8 and k+16+(L>>4)*8 (8 contiguous bf16 each = one b128 load).
__device__ __forceinline__ void load_frag_a(Frag16& fr, const unsigned short* base,
                                            int ld, int m, int k, int lane) {
  const int half = lane >> 4;
  const int lrow = lane & 15;
  const unsigned short* p = base + (size_t)(m + lrow) * ld + k + half * 8;
  fr.u[0] = *(const v4u*)(p);
  fr.u[1] = *(const v4u*)(p + 16);
}

// 16-bit B matrix 32x16 (KxN): lane L -> col N=L&15; K chunks at
// k+(L>>4)*8 and k+16+(L>>4)*8. Weights stored [N,K] row-major.
__device__ __forceinline__ void load_frag_b(Frag16& fr, const unsigned short* base,
                                            int ld, int n, int k, int lane) {
  const int half = lane >> 4;
  const int lcol = lane & 15;
  const unsigned short* p = base + (size_t)(n + lcol) * ld + k + half * 8;
  fr.u[0] = *(const v4u*)(p);
  fr.u[1] = *(const v4u*)(p + 16);
}

__global__ void cvt_f32_bf16(const float* __restrict__ in,
                             unsigned short* __restrict__ out, size_t n) {
  size_t i = (size_t)blockIdx.x * blockDim.x + threadIdx.x;
  size_t stride = (size_t)gridDim.x * blockDim.x;
  for (; i < n; i += stride) out[i] = f32_to_bf16_rne(in[i]);
}

// gates_x[r, n] = x[r,:] @ Wi[n,:] + bi[n] + bh[n]
// Block tile 64(M) x 64(N); 8 waves, each wave owns 2 output tiles sharing A.
// Bandwidth-bound on gx stores; direct global->VGPR fragment loads pipeline well.
__global__ __launch_bounds__(256) void gemm_gates_x(
    const unsigned short* __restrict__ x,    // [65536, 512] bf16
    const unsigned short* __restrict__ wi,   // [2048, 512]  bf16
    const float* __restrict__ bi, const float* __restrict__ bh,
    float* __restrict__ gx)                  // [65536, 2048] fp32
{
  const int lane = threadIdx.x & 31;
  const int wave = threadIdx.x >> 5;
  const int nb = blockIdx.x * 64;
  const int mb = blockIdx.y * 64;
  const int m0 = mb + (wave & 3) * 16;
  const int n0 = nb + ((wave >> 2) * 2) * 16;
  const int n1 = n0 + 16;

  v8f acc0 = {}; v8f acc1 = {};
  for (int k = 0; k < DIN; k += 32) {
    Frag16 a, b0, b1;
    load_frag_a(a,  x,  DIN, m0, k, lane);
    load_frag_b(b0, wi, DIN, n0, k, lane);
    load_frag_b(b1, wi, DIN, n1, k, lane);
    acc0 = __builtin_amdgcn_wmma_f32_16x16x32_bf16(false, a.v, false, b0.v,
                                                   (short)0, acc0, false, false);
    acc1 = __builtin_amdgcn_wmma_f32_16x16x32_bf16(false, a.v, false, b1.v,
                                                   (short)0, acc1, false, false);
  }
  const int half = lane >> 4;
  const int c0 = n0 + (lane & 15);
  const int c1 = n1 + (lane & 15);
  const float bs0 = bi[c0] + bh[c0];
  const float bs1 = bi[c1] + bh[c1];
#pragma unroll
  for (int i = 0; i < 8; ++i) {
    const int row = m0 + half * 8 + i;
    gx[(size_t)row * HGATES + c0] = acc0[i] + bs0;
    gx[(size_t)row * HGATES + c1] = acc1[i] + bs1;
  }
}

// Persistent recurrence. 32 blocks; block b owns h columns [16b, 16b+16) and
// gate columns {16b..} in each of the 4 quadrants (i,f,g,o). Grid barrier per
// timestep; h double-buffered in bf16, c fp32 block-exclusive.
// Wh rows staged to LDS once; h staged to LDS (coalesced) each step; WMMA
// fragments are fed from LDS (ds_load_b128). gx for step t+1 is prefetched.
__global__ __launch_bounds__(256) void lstm_scan(
    const float* __restrict__ gx,            // [64*1024, 2048], row = b*1024+t
    const unsigned short* __restrict__ wh,   // [2048, 512] bf16
    unsigned short* __restrict__ hbuf,       // [2][64*512] bf16 (zero-init)
    float* __restrict__ cstate,              // [64*512] fp32 (zero-init)
    float* __restrict__ out,                 // outputs | h_last | c_last
    unsigned int* __restrict__ barrier_cnt)  // zero-init
{
  __shared__ unsigned short wh_lds[64 * LDSROW];  // block's 64 Wh rows (4 quads x 16)
  __shared__ unsigned short h_lds[64 * LDSROW];   // current h [64 x 512]
  __shared__ float gl[4][4][16 * 17];             // [quadrant][mtile][16x16 padded]

  const int lane  = threadIdx.x & 31;
  const int wave  = threadIdx.x >> 5;
  const int half  = lane >> 4;
  const int lq    = lane & 15;
  const int cbase = blockIdx.x * 16;
  const int q     = wave >> 1;               // quadrant 0..3
  const int mt0   = (wave & 1) * 2;          // m-tile pair {0,1} or {2,3}
  const int mt1   = mt0 + 1;

  const int ecol = threadIdx.x & 15;         // elementwise assignment
  const int erb  = threadIdx.x >> 4;         // 0..15 -> rows erb*4 .. erb*4+3
  const int gcol = cbase + ecol;

  // Stage this block's 64 Wh rows into LDS once (local row r: quad r>>4, col r&15).
  for (int i = threadIdx.x; i < 64 * 64; i += 256) {     // 64 rows x 64 chunks(8 bf16)
    const int r = i >> 6, c = i & 63;
    const int grow = (r >> 4) * H4 + cbase + (r & 15);
    *(v4u*)&wh_lds[r * LDSROW + c * 8] =
        *(const v4u*)(wh + (size_t)grow * H4 + c * 8);
  }
  // (first in-loop __syncthreads covers wh_lds visibility)

  for (int t = 0; t < SEQLEN; ++t) {
    // Prefetch next step's gx lines (one 64B line per thread) to overlap HBM
    // latency with this step's WMMA work.
    if (t + 1 < SEQLEN) {
      const int pr = threadIdx.x >> 2;        // batch row 0..63
      const int pq = threadIdx.x & 3;         // quadrant
      __builtin_prefetch(&gx[((size_t)pr * SEQLEN + (t + 1)) * HGATES + pq * H4 + cbase], 0, 0);
    }

    const unsigned short* hread  = hbuf + (size_t)(t & 1) * (BATCH * H4);
    unsigned short*       hwrite = hbuf + (size_t)((t + 1) & 1) * (BATCH * H4);

    // Cooperatively stage h[64,512] into LDS (coalesced b128 loads).
    for (int i = threadIdx.x; i < 64 * 64; i += 256) {
      const int r = i >> 6, c = i & 63;
      *(v4u*)&h_lds[r * LDSROW + c * 8] =
          *(const v4u*)(hread + (size_t)r * H4 + c * 8);
    }
    __syncthreads();

    v8f acc0 = {}; v8f acc1 = {};
    for (int k = 0; k < H4; k += 32) {
      Frag16 bfr, a0, a1;
      // B frag from wh_lds: lane -> local row q*16 + (lane&15)
      {
        const unsigned short* p = &wh_lds[(q * 16 + lq) * LDSROW + k + half * 8];
        bfr.u[0] = *(const v4u*)(p);
        bfr.u[1] = *(const v4u*)(p + 16);
      }
      // A frags from h_lds
      {
        const unsigned short* p = &h_lds[(mt0 * 16 + lq) * LDSROW + k + half * 8];
        a0.u[0] = *(const v4u*)(p);
        a0.u[1] = *(const v4u*)(p + 16);
      }
      {
        const unsigned short* p = &h_lds[(mt1 * 16 + lq) * LDSROW + k + half * 8];
        a1.u[0] = *(const v4u*)(p);
        a1.u[1] = *(const v4u*)(p + 16);
      }
      acc0 = __builtin_amdgcn_wmma_f32_16x16x32_bf16(false, a0.v, false, bfr.v,
                                                     (short)0, acc0, false, false);
      acc1 = __builtin_amdgcn_wmma_f32_16x16x32_bf16(false, a1.v, false, bfr.v,
                                                     (short)0, acc1, false, false);
    }
#pragma unroll
    for (int i = 0; i < 8; ++i) {
      const int r = half * 8 + i;
      gl[q][mt0][r * 17 + lq] = acc0[i];
      gl[q][mt1][r * 17 + lq] = acc1[i];
    }
    __syncthreads();

#pragma unroll
    for (int rr = 0; rr < 4; ++rr) {
      const int row = erb * 4 + rr;                          // batch row 0..63
      const size_t gxb = ((size_t)row * SEQLEN + t) * HGATES;
      const int lrow = (row & 15) * 17 + ecol;
      const int mti  = row >> 4;
      const float gi = gl[0][mti][lrow] + gx[gxb + 0 * H4 + gcol];
      const float gf = gl[1][mti][lrow] + gx[gxb + 1 * H4 + gcol];
      const float gg = gl[2][mti][lrow] + gx[gxb + 2 * H4 + gcol];
      const float go = gl[3][mti][lrow] + gx[gxb + 3 * H4 + gcol];
      const float si = 1.0f / (1.0f + expf(-gi));
      const float sf = 1.0f / (1.0f + expf(-gf));
      const float tg = tanhf(gg);
      const float so = 1.0f / (1.0f + expf(-go));
      const size_t ci = (size_t)row * H4 + gcol;
      const float cn = sf * cstate[ci] + si * tg;
      const float hn = so * tanhf(cn);
      cstate[ci] = cn;
      hwrite[ci] = f32_to_bf16_rne(hn);
      out[((size_t)row * SEQLEN + t) * H4 + gcol] = hn;
    }

    // grid-wide barrier (monotonic counter; 32 co-resident blocks)
    __syncthreads();
    __threadfence();
    if (threadIdx.x == 0) {
      atomicAdd(barrier_cnt, 1u);
      const unsigned target = 32u * (unsigned)(t + 1);
      while (__hip_atomic_load(barrier_cnt, __ATOMIC_RELAXED,
                               __HIP_MEMORY_SCOPE_AGENT) < target) {
        __builtin_amdgcn_s_sleep(2);
      }
    }
    __syncthreads();
    __threadfence();
  }

  // Tail outputs: h_last = outputs[:, S-1, :], c_last = cstate
  float* hlast = out + (size_t)BATCH * SEQLEN * H4;
  float* clast = hlast + (size_t)BATCH * H4;
#pragma unroll
  for (int rr = 0; rr < 4; ++rr) {
    const int row = erb * 4 + rr;
    const size_t ci = (size_t)row * H4 + gcol;
    hlast[ci] = out[((size_t)row * SEQLEN + (SEQLEN - 1)) * H4 + gcol];
    clast[ci] = cstate[ci];
  }
}

extern "C" void kernel_launch(void* const* d_in, const int* in_sizes, int n_in,
                              void* d_out, int out_size, void* d_ws, size_t ws_size,
                              hipStream_t stream) {
  (void)in_sizes; (void)n_in; (void)out_size; (void)ws_size;
  const float* x_f  = (const float*)d_in[0];   // [64,1024,512]
  const float* wi_f = (const float*)d_in[1];   // [2048,512]
  const float* wh_f = (const float*)d_in[2];   // [2048,512]
  const float* bi   = (const float*)d_in[3];   // [2048]
  const float* bh   = (const float*)d_in[4];   // [2048]
  float* out = (float*)d_out;

  char* ws = (char*)d_ws;
  size_t off = 0;
  unsigned int*  cnt   = (unsigned int*)(ws + off);  off += 256;
  unsigned short* hbuf = (unsigned short*)(ws + off); off += (size_t)2 * BATCH * H4 * 2;      // 128 KB
  float*  cstate       = (float*)(ws + off);          off += (size_t)BATCH * H4 * 4;          // 128 KB
  unsigned short* xb   = (unsigned short*)(ws + off); off += (size_t)NROWS * DIN * 2;         // 64 MB
  unsigned short* wib  = (unsigned short*)(ws + off); off += (size_t)HGATES * DIN * 2;        // 2 MB
  unsigned short* whb  = (unsigned short*)(ws + off); off += (size_t)HGATES * H4 * 2;         // 2 MB
  float* gx            = (float*)(ws + off);          off += (size_t)NROWS * HGATES * 4;      // 512 MB

  // Zero: barrier counter + h double-buffer + c state (bf16/f32 zero == 0x0000)
  hipMemsetAsync(ws, 0, 256 + (size_t)2 * BATCH * H4 * 2 + (size_t)BATCH * H4 * 4, stream);

  cvt_f32_bf16<<<2048, 256, 0, stream>>>(x_f,  xb,  (size_t)NROWS * DIN);
  cvt_f32_bf16<<<512,  256, 0, stream>>>(wi_f, wib, (size_t)HGATES * DIN);
  cvt_f32_bf16<<<512,  256, 0, stream>>>(wh_f, whb, (size_t)HGATES * H4);

  gemm_gates_x<<<dim3(HGATES / 64, NROWS / 64), 256, 0, stream>>>(xb, wib, bi, bh, gx);

  lstm_scan<<<H4 / 16, 256, 0, stream>>>(gx, whb, hbuf, cstate, out, cnt);
}
